// ViG_GNN_35433480192924
// MI455X (gfx1250) — matrix operand order repaired and checked by reading, not compiled, not run on hardware
//
#include <hip/hip_runtime.h>
#include <hip/hip_bf16.h>
#include <math.h>

// ---------------------------------------------------------------------------
// ViG GNN head for MI455X (gfx1250, wave32, WMMA).
//  x[8192,192] --SAGEConv--> h[8192,256] --TopK(0.75)--> hp[6144,256]
//  --kNN(k=9, Gram via WMMA + async-LDS staging, fused top-9)-->
//  SAGEConv --> out[6144,10]
// ---------------------------------------------------------------------------

#define N_NODES 8192
#define D_IN    192
#define D_H     256
#define KNN     9
#define NPOOL   6144
#define NEDGE   (N_NODES * KNN)     // 73728
#define NCLS    10
#define MROWS   128                 // rows per kNN block (8 waves x 16)
#define CTILES  (NPOOL / 16)        // 384 column tiles in Gram
#define TILE_E  (16 * D_H)          // 4096 f16 elements per B tile (8 KB)
#define NEG_INF (-3.0e38f)

typedef _Float16 v16h  __attribute__((ext_vector_type(16)));
typedef _Float16 h2_t  __attribute__((ext_vector_type(2)));
typedef float    v8f   __attribute__((ext_vector_type(8)));
typedef int      v4i_t __attribute__((ext_vector_type(4)));

#if defined(__has_builtin)
#if __has_builtin(__builtin_amdgcn_global_load_async_to_lds_b128) && \
    __has_builtin(__builtin_amdgcn_s_wait_asynccnt)
#define KNN_ASYNC 1
#endif
#endif
#ifndef KNN_ASYNC
#define KNN_ASYNC 0
#endif

// ---- WMMA fragment loaders (CDNA5 ISA 7.12.2 layouts) ----------------------
// A 16x32 f16 (M x K): lane 0-15 -> M=lane, K {0..7}+{16..23}; lane 16-31 ->
// M=lane-16, K {8..15}+{24..31}; 2 f16 per VGPR.
__device__ __forceinline__ v16h load_frag_a(const _Float16* base, int ld,
                                            int row0, int k0) {
  const int lane = threadIdx.x & 31;
  const int hi   = (lane >> 4) & 1;
  const _Float16* p = base + (size_t)(row0 + (lane & 15)) * ld + k0;
  v16h a;
#pragma unroll
  for (int v = 0; v < 8; ++v) {
    const int koff = 2 * (v & 3) + ((v & 4) ? 16 : 0) + hi * 8;
    h2_t q = *(const h2_t*)(p + koff);
    a[2 * v]     = q[0];
    a[2 * v + 1] = q[1];
  }
  return a;
}

// B 32x16 f16 (K x N) from a row-major [N x K] buffer (i.e. B^T):
// lane 0-15 -> N=lane, K {0..7}+{16..23}; lane 16-31 -> N=lane-16,
// K {8..15}+{24..31} (same K order as load_frag_a).
__device__ __forceinline__ v16h load_frag_bt(const _Float16* baseT, int ldk,
                                             int n0, int k0) {
  const int lane = threadIdx.x & 31;
  const int hi   = (lane >> 4) & 1;
  const _Float16* p = baseT + (size_t)(n0 + (lane & 15)) * ldk + k0;
  v16h b;
#pragma unroll
  for (int v = 0; v < 8; ++v) {
    const int koff = 2 * (v & 3) + ((v & 4) ? 16 : 0) + hi * 8;
    h2_t q = *(const h2_t*)(p + koff);
    b[2 * v]     = q[0];
    b[2 * v + 1] = q[1];
  }
  return b;
}

// ---- async global->LDS staging helpers -------------------------------------
// builtin signature (from probe diagnostic): (v4i addrspace(1)*, v4i addrspace(3)*, imm, imm)
__device__ __forceinline__ void copy16_async(const _Float16* g, _Float16* l) {
#if KNN_ASYNC
  v4i_t* gg = (v4i_t*)const_cast<_Float16*>(g);
  v4i_t* ll = (v4i_t*)l;
  __builtin_amdgcn_global_load_async_to_lds_b128(
      (__attribute__((address_space(1))) v4i_t*)gg,
      (__attribute__((address_space(3))) v4i_t*)ll, 0, 0);
#else
  (void)g; (void)l;
#endif
}

// stage one 16x256 f16 tile (rows ct*16 .. ct*16+15 of hph, contiguous 8 KB)
__device__ __forceinline__ void stage_tile_async(const _Float16* hph, int ct,
                                                 _Float16* dst, int t) {
  const _Float16* g = hph + (size_t)ct * TILE_E + t * 8;
  copy16_async(g, dst + t * 8);
  copy16_async(g + 2048, dst + t * 8 + 2048);
}

__device__ __forceinline__ void stage_tile_sync(const _Float16* hph, int ct,
                                                _Float16* dst, int t) {
  const uint4* g = (const uint4*)(hph + (size_t)ct * TILE_E);
  uint4* d = (uint4*)dst;
  d[t]       = g[t];
  d[t + 256] = g[t + 256];
}

// ---------------------------------------------------------------------------
// K0: zero the scatter accumulators (agg[8192*192] ++ cnt[8192])
__global__ void k_zero(float* buf, int n) {
  int i = blockIdx.x * blockDim.x + threadIdx.x;
  if (i < n) buf[i] = 0.0f;
}

// K1: edge scatter: agg[dst,:] += x[src,:], cnt[dst] += 1
__global__ void k_scatter(const float* __restrict__ x,
                          const long long* __restrict__ ei,
                          float* __restrict__ agg, float* __restrict__ cnt) {
  long long t = (long long)blockIdx.x * blockDim.x + threadIdx.x;
  if (t >= (long long)NEDGE * D_IN) return;
  int e = (int)(t / D_IN);
  int f = (int)(t % D_IN);
  int s = (int)ei[e];
  int d = (int)ei[NEDGE + e];
  atomicAdd(&agg[(size_t)d * D_IN + f], x[(size_t)s * D_IN + f]);
  if (f == 0) atomicAdd(&cnt[d], 1.0f);
}

// K2: normalize agg by count, convert agg and x to f16
__global__ void k_prep_af16(const float* __restrict__ agg,
                            const float* __restrict__ cnt,
                            const float* __restrict__ x,
                            _Float16* __restrict__ aggh,
                            _Float16* __restrict__ xh) {
  int i = blockIdx.x * blockDim.x + threadIdx.x;
  if (i >= N_NODES * D_IN) return;
  int row = i / D_IN;
  float c = cnt[row];
  float inv = 1.0f / fmaxf(c, 1.0f);
  aggh[i] = (_Float16)(agg[i] * inv);
  xh[i]   = (_Float16)x[i];
}

// K3: transpose + convert stage-1 weights [192,256] -> [256,192] f16
__global__ void k_prep_w(const float* __restrict__ Wl, const float* __restrict__ Wr,
                         _Float16* __restrict__ WlT, _Float16* __restrict__ WrT) {
  int i = blockIdx.x * blockDim.x + threadIdx.x;
  if (i >= D_IN * D_H) return;
  int n = i % D_H;
  int k = i / D_H;
  WlT[(size_t)n * D_IN + k] = (_Float16)Wl[(size_t)k * D_H + n];
  WrT[(size_t)n * D_IN + k] = (_Float16)Wr[(size_t)k * D_H + n];
}

// K4: h = mean_agg @ Wl1 + b + x @ Wr1  (WMMA, one 16x16 C tile per wave)
__global__ void __launch_bounds__(256)
k_gemm1(const _Float16* __restrict__ aggh, const _Float16* __restrict__ xh,
        const _Float16* __restrict__ WlT,  const _Float16* __restrict__ WrT,
        const float* __restrict__ bias, float* __restrict__ h) {
  const int lane = threadIdx.x & 31;
  const int wave = threadIdx.x >> 5;
  const int tile = blockIdx.x * 8 + wave;            // 8192 tiles
  const int tm = tile >> 4;
  const int tn = tile & 15;
  const int m0 = tm * 16, n0 = tn * 16;
  const int col = n0 + (lane & 15);

  v8f acc;
  const float bv = bias[col];
#pragma unroll
  for (int r = 0; r < 8; ++r) acc[r] = bv;

#pragma unroll
  for (int ks = 0; ks < D_IN / 32; ++ks) {
    v16h a = load_frag_a(aggh, D_IN, m0, ks * 32);
    v16h b = load_frag_bt(WlT, D_IN, n0, ks * 32);
    acc = __builtin_amdgcn_wmma_f32_16x16x32_f16(false, a, false, b,
                                                 (short)0, acc, false, false);
  }
#pragma unroll
  for (int ks = 0; ks < D_IN / 32; ++ks) {
    v16h a = load_frag_a(xh, D_IN, m0, ks * 32);
    v16h b = load_frag_bt(WrT, D_IN, n0, ks * 32);
    acc = __builtin_amdgcn_wmma_f32_16x16x32_f16(false, a, false, b,
                                                 (short)0, acc, false, false);
  }

  const int hi = (lane >> 4) & 1;
#pragma unroll
  for (int r = 0; r < 8; ++r) {
    int m = m0 + r + hi * 8;
    h[(size_t)m * D_H + col] = acc[r];
  }
}

// K5: score = tanh(h . w / ||w||)
__global__ void k_score(const float* __restrict__ h, const float* __restrict__ w,
                        float* __restrict__ score) {
  int i = blockIdx.x * blockDim.x + threadIdx.x;
  if (i >= N_NODES) return;
  float wn = 0.0f, dot = 0.0f;
  for (int k = 0; k < D_H; ++k) {
    float wk = w[k];
    wn += wk * wk;
    dot += h[(size_t)i * D_H + k] * wk;
  }
  score[i] = tanhf(dot * rsqrtf(wn));
}

// K6: exact top-k by rank counting (tie-break identical to jax.lax.top_k)
__global__ void __launch_bounds__(256)
k_rank_select(const float* __restrict__ score, int* __restrict__ sel,
              float* __restrict__ tops) {
  __shared__ int sh[256];
  const int i = blockIdx.x;
  const float si = score[i];
  int c = 0;
  for (int j = threadIdx.x; j < N_NODES; j += 256) {
    float sj = score[j];
    c += (sj > si) || (sj == si && j < i);
  }
  sh[threadIdx.x] = c;
  __syncthreads();
  for (int s = 128; s > 0; s >>= 1) {
    if (threadIdx.x < s) sh[threadIdx.x] += sh[threadIdx.x + s];
    __syncthreads();
  }
  if (threadIdx.x == 0) {
    int rank = sh[0];
    if (rank < NPOOL) { sel[rank] = i; tops[rank] = si; }
  }
}

// K7: hp = h[perm] * top_s ; also f16 copy
__global__ void k_build_hp(const float* __restrict__ h, const int* __restrict__ sel,
                           const float* __restrict__ tops,
                           float* __restrict__ hp, _Float16* __restrict__ hph) {
  int r = blockIdx.x;
  int k = threadIdx.x;
  float v = h[(size_t)sel[r] * D_H + k] * tops[r];
  hp[(size_t)r * D_H + k] = v;
  hph[(size_t)r * D_H + k] = (_Float16)v;
}

// K8: squared norms of pooled rows
__global__ void k_sqn(const float* __restrict__ hp, float* __restrict__ sqn) {
  int r = blockIdx.x * blockDim.x + threadIdx.x;
  if (r >= NPOOL) return;
  float s = 0.0f;
  for (int k = 0; k < D_H; ++k) {
    float v = hp[(size_t)r * D_H + k];
    s += v * v;
  }
  sqn[r] = s;
}

// K9: fused Gram + top-9.  s = 2*dot - |hp_j|^2 (order == -distance per row).
// Block = 128 rows; wave w owns rows [w*16, w*16+16) with its A fragments
// resident in 64 VGPRs.  All waves share one 16x256 B tile staged in LDS,
// double-buffered via GLOBAL_LOAD_ASYNC_TO_LDS_B128 (ASYNCcnt) when available.
__global__ void __launch_bounds__(256)
k_knn(const _Float16* __restrict__ hph, const float* __restrict__ sqn,
      int* __restrict__ nbr) {
  __shared__ _Float16 sB[2][TILE_E];       // 2 x 8 KB B tiles
  __shared__ float s_tile[8][16][16];      // per-wave score staging

  const int lane = threadIdx.x & 31;
  const int wave = threadIdx.x >> 5;
  const int hi   = (lane >> 4) & 1;
  const int t    = threadIdx.x;
  const int m0   = blockIdx.x * MROWS + wave * 16;

  // A fragments for the wave's 16 rows, K = 0..255 -- loaded once.
  v16h afrag[8];
#pragma unroll
  for (int ks = 0; ks < 8; ++ks)
    afrag[ks] = load_frag_a(hph, D_H, m0, ks * 32);

  float bs[KNN];
  int   bj[KNN];
#pragma unroll
  for (int q = 0; q < KNN; ++q) { bs[q] = NEG_INF; bj[q] = 0x7fffffff; }

#if KNN_ASYNC
  stage_tile_async(hph, 0, sB[0], t);
  __builtin_amdgcn_s_wait_asynccnt(0);
  __syncthreads();
#endif

  for (int ct = 0; ct < CTILES; ++ct) {
#if KNN_ASYNC
    const _Float16* buf = sB[ct & 1];
    if (ct + 1 < CTILES) stage_tile_async(hph, ct + 1, sB[(ct + 1) & 1], t);
#else
    const _Float16* buf = sB[0];
    __syncthreads();
    stage_tile_sync(hph, ct, sB[0], t);
    __syncthreads();
#endif

    v8f acc = {};
#pragma unroll
    for (int ks = 0; ks < 8; ++ks) {
      v16h b = load_frag_bt(buf, D_H, 0, ks * 32);   // ds_load from LDS tile
      acc = __builtin_amdgcn_wmma_f32_16x16x32_f16(false, afrag[ks], false, b,
                                                   (short)0, acc, false, false);
    }

    const int cn = lane & 15;
    const float sq_j = sqn[ct * 16 + cn];
#pragma unroll
    for (int r = 0; r < 8; ++r)
      s_tile[wave][r + hi * 8][cn] = 2.0f * acc[r] - sq_j;
    __syncthreads();   // uniform: publishes s_tile (and fences LDS per wave)

    if (lane < 16) {
      const int row = lane;
#pragma unroll
      for (int c = 0; c < 16; ++c) {
        float s = s_tile[wave][row][c];
        int   j = ct * 16 + c;
        bool better = (s > bs[KNN - 1]) ||
                      (s == bs[KNN - 1] && j < bj[KNN - 1]);
        if (better) {
          bs[KNN - 1] = s; bj[KNN - 1] = j;
#pragma unroll
          for (int q = KNN - 1; q > 0; --q) {
            bool sw = (bs[q] > bs[q - 1]) ||
                      (bs[q] == bs[q - 1] && bj[q] < bj[q - 1]);
            if (sw) {
              float tf = bs[q]; bs[q] = bs[q - 1]; bs[q - 1] = tf;
              int   ti = bj[q]; bj[q] = bj[q - 1]; bj[q - 1] = ti;
            }
          }
        }
      }
    }

#if KNN_ASYNC
    __builtin_amdgcn_s_wait_asynccnt(0);   // next tile resident in LDS
#endif
    __syncthreads();                       // safe to recycle buffers
  }

  // each wave owns its rows outright -> no cross-wave merge needed
  if (lane < 16) {
#pragma unroll
    for (int q = 0; q < KNN; ++q)
      nbr[(size_t)(m0 + lane) * KNN + q] = bj[q];
  }
}

// K10: out = mean(hp[nbr]) @ Wl2 + b + hp @ Wr2   (one block per pooled row)
__global__ void __launch_bounds__(256)
k_final(const float* __restrict__ hp, const int* __restrict__ nbr,
        const float* __restrict__ Wl2, const float* __restrict__ bl2,
        const float* __restrict__ Wr2, float* __restrict__ out) {
  __shared__ float sAgg[D_H];
  __shared__ float sHp[D_H];
  const int i = blockIdx.x;
  const int k = threadIdx.x;
  float a = 0.0f;
#pragma unroll
  for (int q = 0; q < KNN; ++q) {
    int j = nbr[(size_t)i * KNN + q];
    a += hp[(size_t)j * D_H + k];
  }
  sAgg[k] = a * (1.0f / (float)KNN);
  sHp[k]  = hp[(size_t)i * D_H + k];
  __syncthreads();
  if (k < NCLS) {
    float acc = bl2[k];
    for (int kk = 0; kk < D_H; ++kk)
      acc += sAgg[kk] * Wl2[(size_t)kk * NCLS + k] +
             sHp[kk]  * Wr2[(size_t)kk * NCLS + k];
    out[(size_t)i * NCLS + k] = acc;
  }
}

// ---------------------------------------------------------------------------
extern "C" void kernel_launch(void* const* d_in, const int* in_sizes, int n_in,
                              void* d_out, int out_size, void* d_ws, size_t ws_size,
                              hipStream_t stream) {
  (void)in_sizes; (void)n_in; (void)out_size; (void)ws_size;
  const float*     x      = (const float*)d_in[0];
  const long long* ei     = (const long long*)d_in[1];   // int64 [2, E]
  const float*     W_l1   = (const float*)d_in[2];
  const float*     b_l1   = (const float*)d_in[3];
  const float*     W_r1   = (const float*)d_in[4];
  const float*     pool_w = (const float*)d_in[5];
  const float*     W_l2   = (const float*)d_in[6];
  const float*     b_l2   = (const float*)d_in[7];
  const float*     W_r2   = (const float*)d_in[8];
  float* out = (float*)d_out;

  char* ws = (char*)d_ws;
  size_t off = 0;
  auto carve = [&](size_t bytes) -> char* {
    char* p = ws + off;
    off = (off + bytes + 255) & ~(size_t)255;
    return p;
  };
  float*    agg  = (float*)carve((size_t)N_NODES * D_IN * 4);
  float*    cnt  = (float*)carve((size_t)N_NODES * 4);
  float*    h    = (float*)carve((size_t)N_NODES * D_H * 4);
  _Float16* aggh = (_Float16*)carve((size_t)N_NODES * D_IN * 2);
  _Float16* xh   = (_Float16*)carve((size_t)N_NODES * D_IN * 2);
  _Float16* WlT  = (_Float16*)carve((size_t)D_IN * D_H * 2);
  _Float16* WrT  = (_Float16*)carve((size_t)D_IN * D_H * 2);
  float*    scr  = (float*)carve((size_t)N_NODES * 4);
  int*      sel  = (int*)carve((size_t)NPOOL * 4);
  float*    tops = (float*)carve((size_t)NPOOL * 4);
  float*    hp   = (float*)carve((size_t)NPOOL * D_H * 4);
  _Float16* hph  = (_Float16*)carve((size_t)NPOOL * D_H * 2);
  float*    sqn  = (float*)carve((size_t)NPOOL * 4);
  int*      nbr  = (int*)carve((size_t)NPOOL * KNN * 4);

  const int zn = N_NODES * D_IN + N_NODES;   // agg ++ cnt are contiguous
  k_zero<<<(zn + 255) / 256, 256, 0, stream>>>(agg, zn);

  const long long st = (long long)NEDGE * D_IN;
  k_scatter<<<(unsigned)((st + 255) / 256), 256, 0, stream>>>(x, ei, agg, cnt);

  k_prep_af16<<<(N_NODES * D_IN + 255) / 256, 256, 0, stream>>>(agg, cnt, x, aggh, xh);
  k_prep_w<<<(D_IN * D_H + 255) / 256, 256, 0, stream>>>(W_l1, W_r1, WlT, WrT);

  k_gemm1<<<(N_NODES / 16) * (D_H / 16) / 8, 256, 0, stream>>>(aggh, xh, WlT, WrT,
                                                               b_l1, h);
  k_score<<<(N_NODES + 255) / 256, 256, 0, stream>>>(h, pool_w, scr);
  k_rank_select<<<N_NODES, 256, 0, stream>>>(scr, sel, tops);
  k_build_hp<<<NPOOL, D_H, 0, stream>>>(h, sel, tops, hp, hph);
  k_sqn<<<(NPOOL + 255) / 256, 256, 0, stream>>>(hp, sqn);
  k_knn<<<NPOOL / MROWS, 256, 0, stream>>>(hph, sqn, nbr);
  k_final<<<NPOOL, D_H, 0, stream>>>(hp, nbr, W_l2, b_l2, W_r2, out);
}
